// LocalRefineBlock_231928234021
// MI455X (gfx1250) — matrix operand order repaired and checked by reading, not compiled, not run on hardware
//
#include <hip/hip_runtime.h>

// ---------------------------------------------------------------------------
// LocalRefineBlock forward for MI455X (gfx1250, wave32, WMMA + TDM).
// GEMM phase: bf16 operands, f32 accumulate via v_wmma_f32_16x16x32_bf16,
// 4 accumulators per wave (A fragment reused 4x); B fragments for all 4
// N-tiles are loaded up-front so loads clause together and overlap WMMAs.
// Heavy 3x3 conv streams the context K-slice into LDS via tensor_load_to_lds
// (double buffered, s_wait_tensorcnt + barrier handoff); all 9 taps read A
// from LDS. GEMM working set (context 33MB + weights 36MB) is L2-resident
// (192MB); only the 2x168MB global_x writeback is HBM-bound (~14us @23.3TB/s).
// ---------------------------------------------------------------------------

#define BB   32
#define CC   320
#define CTXC 1024
#define PP   256     // context positions 16x16
#define RR   81      // roi positions 9x9
#define RP   96      // padded roi rows (6 WMMA tiles)
#define HW   4096    // 64x64

typedef __bf16 bf16_t;
typedef __attribute__((ext_vector_type(16))) __bf16 v16bf;
typedef __attribute__((ext_vector_type(8)))  __bf16 v8bf;
typedef __attribute__((ext_vector_type(8)))  float  v8f;
typedef __attribute__((ext_vector_type(4)))  unsigned int u32x4;
typedef __attribute__((ext_vector_type(8)))  int i32x8;
typedef __attribute__((ext_vector_type(4)))  int i32x4;

__device__ inline v16bf frag_zero() {
  v16bf z;
#pragma unroll
  for (int i = 0; i < 16; ++i) z[i] = (bf16_t)0.0f;
  return z;
}
// A fragment (16x32 bf16): lane holds row (lane&15); elems 0..7 = K[hi*8..],
// elems 8..15 = K[16+hi*8..]  (hi = lane>>4)
__device__ inline v16bf loadA(const bf16_t* row_k0, int hi) {
  union { v16bf v; v8bf h[2]; } u;
  u.h[0] = *(const v8bf*)(row_k0 + hi * 8);
  u.h[1] = *(const v8bf*)(row_k0 + 16 + hi * 8);
  return u.v;
}
// B fragment (32x16 bf16): lane holds column (lane&15); elems 0..7 = K[hi*8..],
// elems 8..15 = K[16+hi*8..]  (hi = lane>>4)
__device__ inline v16bf loadB(const bf16_t* col_k0, int hi) {
  union { v16bf v; v8bf h[2]; } u;
  u.h[0] = *(const v8bf*)(col_k0 + hi * 8);
  u.h[1] = *(const v8bf*)(col_k0 + 16 + hi * 8);
  return u.v;
}
__device__ inline v8f acc_zero() {
  v8f a;
#pragma unroll
  for (int i = 0; i < 8; ++i) a[i] = 0.0f;
  return a;
}
#define WMMA_BF16(A, Bm, Cv) \
  __builtin_amdgcn_wmma_f32_16x16x32_bf16(false, (A), false, (Bm), (short)0, (Cv), false, false)

// TDM: load a 2D bf16 tile [256 rows x 32] (row pitch 1024 elems) into LDS.
// D# per CDNA5 ISA 8.3/8.4: group0 = {count, lds_addr, global_addr, type=2},
// group1 = {data_size=2B, tensor_dim0/1, tile_dim0/1, dim0_stride}.
// This toolchain's builtin takes 6 args (4 descriptor groups + pad + cpol).
__device__ inline void tdm_load_slice(const bf16_t* gptr, unsigned lds_off) {
  unsigned long long ga = (unsigned long long)(size_t)gptr;
  u32x4 g0; i32x8 g1; i32x4 g2; i32x4 g3; i32x8 g4;
  g0[0] = 1u;                                   // count = 1 (valid descriptor)
  g0[1] = lds_off;                              // lds_addr (bytes)
  g0[2] = (unsigned)ga;                         // global_addr[31:0]
  g0[3] = (unsigned)((ga >> 32) & 0x01FFFFFFu)  // global_addr[56:32]
          | (2u << 30);                         // type = 2 ("image")
  g1[0] = (int)0x00010000u;                     // data_size = 2 bytes
  g1[1] = (int)(32u << 16);                     // tensor_dim0 = 32 (bits 79:48 lo)
  g1[2] = (int)(256u << 16);                    // tensor_dim1 = 256 (bits 111:80 lo)
  g1[3] = (int)(32u << 16);                     // tile_dim0 = 32 (bits 127:112)
  g1[4] = (int)256u;                            // tile_dim1 = 256, tile_dim2 = 0
  g1[5] = (int)1024u;                           // tensor_dim0_stride = 1024
  g1[6] = 0; g1[7] = 0;                         // dim1_stride = 0 (2D)
  g2[0] = 0; g2[1] = 0; g2[2] = 0; g2[3] = 0;
  g3[0] = 0; g3[1] = 0; g3[2] = 0; g3[3] = 0;
  g4[0] = 0; g4[1] = 0; g4[2] = 0; g4[3] = 0;
  g4[4] = 0; g4[5] = 0; g4[6] = 0; g4[7] = 0;
  __builtin_amdgcn_tensor_load_to_lds(g0, g1, g2, g3, g4, 0);
}

// ---------------------------------------------------------------------------
// Kernel 0: repack weights to bf16; conv weights reordered to [tap][o][c]
// ---------------------------------------------------------------------------
__global__ __launch_bounds__(256) void k_repack(
    const float* w_ctx, const float* w_out, const float* w_sg, const float* w_sb,
    bf16_t* w_ctx_bf, bf16_t* wt_out, bf16_t* wt_sg, bf16_t* wt_sb) {
  int idx = blockIdx.x * 256 + threadIdx.x;           // over 320*1024
  if (idx >= CC * CTXC) return;
  int o = idx / CTXC, c = idx % CTXC;
  w_ctx_bf[idx] = (bf16_t)w_ctx[idx];
#pragma unroll
  for (int t = 0; t < 9; ++t) {
    size_t dst = ((size_t)t * CC + o) * CTXC + c;
    size_t src = (size_t)idx * 9 + t;                 // (o, c, 3, 3)
    wt_out[dst] = (bf16_t)w_out[src];
    wt_sg[dst]  = (bf16_t)w_sg[src];
    wt_sb[dst]  = (bf16_t)w_sb[src];
  }
}

// ---------------------------------------------------------------------------
// Kernel 1: context groupnorm (32 groups x 32 ch, eps 1e-6) + bf16 packs
// ---------------------------------------------------------------------------
__global__ __launch_bounds__(256) void k_ctxnorm(
    const float* context, const float* cn_gamma, const float* cn_beta,
    bf16_t* ctx_bf, bf16_t* ctxT_bf, bf16_t* kn_bf) {
  int b = blockIdx.x >> 5, g = blockIdx.x & 31, tid = threadIdx.x;
  __shared__ float s1[256], s2[256];
  float su = 0.f, sq = 0.f;
  for (int i = tid; i < 32 * PP; i += 256) {
    int cc2 = i >> 8, p = i & 255;
    float v = context[((size_t)b * PP + p) * CTXC + g * 32 + cc2];
    su += v; sq += v * v;
  }
  s1[tid] = su; s2[tid] = sq; __syncthreads();
  for (int off = 128; off > 0; off >>= 1) {
    if (tid < off) { s1[tid] += s1[tid + off]; s2[tid] += s2[tid + off]; }
    __syncthreads();
  }
  float mu = s1[0] / (32.0f * PP);
  float var = s2[0] / (32.0f * PP) - mu * mu;
  float rs = rsqrtf(var + 1e-6f);
  for (int i = tid; i < 32 * PP; i += 256) {
    int cc2 = i >> 8, p = i & 255;
    int c = g * 32 + cc2;
    float v = context[((size_t)b * PP + p) * CTXC + c];
    bf16_t bv = (bf16_t)v;
    ctx_bf [((size_t)b * PP + p) * CTXC + c] = bv;
    ctxT_bf[((size_t)b * CTXC + c) * PP + p] = bv;
    kn_bf  [((size_t)b * PP + p) * CTXC + c] =
        (bf16_t)((v - mu) * rs * cn_gamma[c] + cn_beta[c]);
  }
}

// ---------------------------------------------------------------------------
// Kernel 2: ROI align -> xroi f32 [b][c][81]
// ---------------------------------------------------------------------------
__global__ __launch_bounds__(256) void k_roialign(
    const float* gx, const float* bbox, float* xroi) {
  int b = blockIdx.x / 102;
  int idx = (blockIdx.x % 102) * 256 + threadIdx.x;
  if (idx >= CC * RR) return;
  int c = idx / RR, p = idx % RR;
  int py = p / 9, px = p % 9;
  float x1 = bbox[b * 4 + 0] * 64.f, y1 = bbox[b * 4 + 1] * 64.f;
  float x2 = bbox[b * 4 + 2] * 64.f, y2 = bbox[b * 4 + 3] * 64.f;
  float rw = fmaxf(x2 - x1, 1.f), rh = fmaxf(y2 - y1, 1.f);
  float xs = x1 + ((float)px + 0.5f) * (rw / 9.f);
  float ys = y1 + ((float)py + 0.5f) * (rh / 9.f);
  bool valid = (xs > -1.f) && (xs < 64.f) && (ys > -1.f) && (ys < 64.f);
  float xc = fminf(fmaxf(xs, 0.f), 63.f);
  float yc = fminf(fmaxf(ys, 0.f), 63.f);
  int x0 = (int)floorf(xc), y0 = (int)floorf(yc);
  int x1i = min(x0 + 1, 63), y1i = min(y0 + 1, 63);
  float lx = xc - (float)x0, ly = yc - (float)y0;
  const float* img = gx + ((size_t)b * CC + c) * HW;
  float v00 = img[y0 * 64 + x0],  v01 = img[y0 * 64 + x1i];
  float v10 = img[y1i * 64 + x0], v11 = img[y1i * 64 + x1i];
  float val = v00 * (1.f - ly) * (1.f - lx) + v01 * (1.f - ly) * lx +
              v10 * ly * (1.f - lx)         + v11 * ly * lx;
  xroi[((size_t)b * CC + c) * RR + p] = valid ? val : 0.f;
}

// ---------------------------------------------------------------------------
// Kernel 3: ln groupnorm stats (32 groups x 10 ch over 81 pos, eps 1e-6)
// ---------------------------------------------------------------------------
__global__ __launch_bounds__(256) void k_lnstats(const float* xroi, float* gnA) {
  int b = blockIdx.x >> 5, g = blockIdx.x & 31, tid = threadIdx.x;
  __shared__ float s1[256], s2[256];
  float su = 0.f, sq = 0.f;
  for (int i = tid; i < 10 * RR; i += 256) {
    int c = g * 10 + i / RR, p = i % RR;
    float v = xroi[((size_t)b * CC + c) * RR + p];
    su += v; sq += v * v;
  }
  s1[tid] = su; s2[tid] = sq; __syncthreads();
  for (int off = 128; off > 0; off >>= 1) {
    if (tid < off) { s1[tid] += s1[tid + off]; s2[tid] += s2[tid + off]; }
    __syncthreads();
  }
  if (tid == 0) {
    float mu = s1[0] / 810.f;
    float var = s2[0] / 810.f - mu * mu;
    gnA[(b * 32 + g) * 2 + 0] = mu;
    gnA[(b * 32 + g) * 2 + 1] = rsqrtf(var + 1e-6f);
  }
}

// ---------------------------------------------------------------------------
// Kernel 4: q = w_in * [gn(xroi); indicator] + b_in  -> q_bf [b][96][320]
// ---------------------------------------------------------------------------
__global__ __launch_bounds__(256) void k_q(
    const float* xroi, const float* gnA, const float* ln_gamma, const float* ln_beta,
    const float* w_in, const float* b_in, const float* indicator, bf16_t* q_bf) {
  int b = blockIdx.x / 120;
  int idx = (blockIdx.x % 120) * 256 + threadIdx.x;   // 96*320 exact
  int p = idx / CC, o = idx % CC;
  if (p >= RR) { q_bf[((size_t)b * RP + p) * CC + o] = (bf16_t)0.f; return; }
  float acc = b_in[o] + w_in[o * 322 + 320] * indicator[b * 2 + 0]
                      + w_in[o * 322 + 321] * indicator[b * 2 + 1];
  for (int i = 0; i < CC; ++i) {
    int g = i / 10;
    float mu = gnA[(b * 32 + g) * 2 + 0];
    float rs = gnA[(b * 32 + g) * 2 + 1];
    float xn = (xroi[((size_t)b * CC + i) * RR + p] - mu) * rs * ln_gamma[i] + ln_beta[i];
    acc += w_in[o * 322 + i] * xn;
  }
  q_bf[((size_t)b * RP + p) * CC + o] = (bf16_t)acc;
}

// ---------------------------------------------------------------------------
// Kernel 5: k = gn(ctx) @ w_ctx^T + b_ctx  (M=256, N=320, K=1024) -> k_bf (p,o)
// One wave = 1 p-tile x 4 n-tiles; B fragments prefetched into an array.
// ---------------------------------------------------------------------------
__global__ __launch_bounds__(256) void k_kgemm(
    const bf16_t* kn_bf, const bf16_t* w_ctx_bf, const float* b_ctx, bf16_t* k_bf) {
  int wid = blockIdx.x * 8 + (threadIdx.x >> 5);
  if (wid >= BB * 16 * 5) return;
  int lane = threadIdx.x & 31, col = lane & 15, hi = lane >> 4;
  int b = wid / 80, r = wid % 80, pt = r / 5, ng = r % 5;
  const bf16_t* Arow = kn_bf + ((size_t)b * PP + pt * 16 + col) * CTXC;
  const bf16_t* Bcol = w_ctx_bf + ((size_t)(ng * 64 + col)) * CTXC;
  v8f acc[4];
#pragma unroll
  for (int nt = 0; nt < 4; ++nt) acc[nt] = acc_zero();
#pragma unroll 2
  for (int k0 = 0; k0 < CTXC; k0 += 32) {
    v16bf a = loadA(Arow + k0, hi);
    v16bf bm[4];
#pragma unroll
    for (int nt = 0; nt < 4; ++nt)
      bm[nt] = loadB(Bcol + (size_t)nt * 16 * CTXC + k0, hi);
#pragma unroll
    for (int nt = 0; nt < 4; ++nt)
      acc[nt] = WMMA_BF16(a, bm[nt], acc[nt]);
  }
  int pbase = pt * 16 + hi * 8;
#pragma unroll
  for (int nt = 0; nt < 4; ++nt) {
    int o = ng * 64 + nt * 16 + col;
    float bias = b_ctx[o];
#pragma unroll
    for (int i = 0; i < 8; ++i)
      k_bf[((size_t)b * PP + pbase + i) * CC + o] = (bf16_t)(acc[nt][i] + bias);
  }
}

// ---------------------------------------------------------------------------
// Kernel 6: v = conv3x3(ctx_map, w_out) -> v_bf (o,p)
// Block = one (b, 64-col group); TDM streams 16KB context K-slices into LDS
// (double buffered); each wave holds 2 p-tiles x 4 n-tiles = 8 accumulators
// and runs all 9 taps out of the LDS slab (72 WMMAs per slice per wave).
// ---------------------------------------------------------------------------
__global__ __launch_bounds__(256) void k_vgemm(
    const bf16_t* ctx_bf, const bf16_t* wt_out, const float* b_out, bf16_t* v_bf) {
  __shared__ bf16_t smem[2][PP * 32];                 // 2 x 16KB K-slices
  int b = blockIdx.x / 5, ng = blockIdx.x % 5;        // 64 output columns
  int wave = threadIdx.x >> 5;
  int lane = threadIdx.x & 31, col = lane & 15, hi = lane >> 4;
  int og = ng * 64;
  const bf16_t* gctx = ctx_bf + (size_t)b * PP * CTXC;
  unsigned lds_base = (unsigned)(unsigned long long)(size_t)&smem[0][0];

  v8f acc[2][4];
#pragma unroll
  for (int s2 = 0; s2 < 2; ++s2)
#pragma unroll
    for (int nt = 0; nt < 4; ++nt) acc[s2][nt] = acc_zero();

  if (wave == 0) {
    tdm_load_slice(gctx, lds_base);
    __builtin_amdgcn_s_wait_tensorcnt(0);
  }
  __syncthreads();

  for (int s = 0; s < 32; ++s) {                      // 32 K-slices of 32
    int cur = s & 1;
    if (wave == 0 && s + 1 < 32)
      tdm_load_slice(gctx + (size_t)(s + 1) * 32,
                     lds_base + (unsigned)((cur ^ 1) * (PP * 32 * 2)));
    for (int t = 0; t < 9; ++t) {
      int dy = t / 3 - 1, dx = t % 3 - 1;
      v16bf bf[4];
#pragma unroll
      for (int nt = 0; nt < 4; ++nt)
        bf[nt] = loadB(wt_out + ((size_t)t * CC + og + nt * 16 + col) * CTXC + s * 32, hi);
#pragma unroll
      for (int sub = 0; sub < 2; ++sub) {
        int prow = (wave * 2 + sub) * 16 + col;
        int y = prow >> 4, x = prow & 15;
        int yy = y + dy, xx = x + dx;
        bool valid = (yy >= 0) && (yy < 16) && (xx >= 0) && (xx < 16);
        v16bf a;
        if (valid) {
          union { v16bf v; v8bf h[2]; } u;
          int ro = (yy * 16 + xx) * 32;
          u.h[0] = *(const v8bf*)&smem[cur][ro + hi * 8];
          u.h[1] = *(const v8bf*)&smem[cur][ro + 16 + hi * 8];
          a = u.v;
        } else {
          a = frag_zero();
        }
#pragma unroll
        for (int nt = 0; nt < 4; ++nt)
          acc[sub][nt] = WMMA_BF16(a, bf[nt], acc[sub][nt]);
      }
    }
    if (wave == 0 && s + 1 < 32) __builtin_amdgcn_s_wait_tensorcnt(0);
    __syncthreads();
  }

#pragma unroll
  for (int sub = 0; sub < 2; ++sub) {
    int pbase = (wave * 2 + sub) * 16 + hi * 8;
#pragma unroll
    for (int nt = 0; nt < 4; ++nt) {
      int o = og + nt * 16 + col;
      float bias = b_out[o];
      v8bf pk;
#pragma unroll
      for (int i = 0; i < 8; ++i) pk[i] = (bf16_t)(acc[sub][nt][i] + bias);
      *(v8bf*)(v_bf + ((size_t)b * CC + o) * PP + pbase) = pk;
    }
  }
}

// ---------------------------------------------------------------------------
// Kernel 7: sim = q @ k^T  (M=96, N=256, K=320), 4 accumulators/wave
// ---------------------------------------------------------------------------
__global__ __launch_bounds__(256) void k_simgemm(
    const bf16_t* q_bf, const bf16_t* k_bf, float* sim_f) {
  int wid = blockIdx.x * 8 + (threadIdx.x >> 5);
  if (wid >= BB * 6 * 4) return;
  int lane = threadIdx.x & 31, col = lane & 15, hi = lane >> 4;
  int b = wid / 24, r = wid % 24, mt = r / 4, ng = r % 4;
  const bf16_t* Arow = q_bf + ((size_t)b * RP + mt * 16 + col) * CC;
  const bf16_t* Bcol = k_bf + ((size_t)b * PP + ng * 64 + col) * CC;
  v8f acc[4];
#pragma unroll
  for (int nt = 0; nt < 4; ++nt) acc[nt] = acc_zero();
#pragma unroll 2
  for (int k0 = 0; k0 < CC; k0 += 32) {
    v16bf a = loadA(Arow + k0, hi);
    v16bf bm[4];
#pragma unroll
    for (int nt = 0; nt < 4; ++nt)
      bm[nt] = loadB(Bcol + (size_t)nt * 16 * CC + k0, hi);
#pragma unroll
    for (int nt = 0; nt < 4; ++nt)
      acc[nt] = WMMA_BF16(a, bm[nt], acc[nt]);
  }
#pragma unroll
  for (int nt = 0; nt < 4; ++nt)
#pragma unroll
    for (int i = 0; i < 8; ++i)
      sim_f[((size_t)b * RP + mt * 16 + hi * 8 + i) * PP + ng * 64 + nt * 16 + col] =
          acc[nt][i];
}

// ---------------------------------------------------------------------------
// Kernel 8: softmax rows + hard-attention mask projection
// ---------------------------------------------------------------------------
__global__ __launch_bounds__(256) void k_softmax(
    const float* sim_f, const float* mask, bf16_t* attn_bf, float* out_mask) {
  int b = blockIdx.x / RP, p = blockIdx.x % RP, j = threadIdx.x;
  bf16_t* arow = attn_bf + ((size_t)b * RP + p) * PP;
  if (p >= RR) { arow[j] = (bf16_t)0.f; return; }
  __shared__ float sm[256];
  float v = sim_f[((size_t)b * RP + p) * PP + j];
  sm[j] = v; __syncthreads();
  for (int off = 128; off > 0; off >>= 1) {
    if (j < off) sm[j] = fmaxf(sm[j], sm[j + off]);
    __syncthreads();
  }
  float m = sm[0]; __syncthreads();
  float e = __expf(v - m);
  sm[j] = e; __syncthreads();
  for (int off = 128; off > 0; off >>= 1) {
    if (j < off) sm[j] += sm[j + off];
    __syncthreads();
  }
  float s = sm[0]; __syncthreads();
  arow[j] = (bf16_t)(e / s);
  sm[j] = (v == m) ? mask[b * PP + j] : 0.f; __syncthreads();
  for (int off = 128; off > 0; off >>= 1) {
    if (j < off) sm[j] += sm[j + off];
    __syncthreads();
  }
  if (j == 0) out_mask[b * RR + p] = fminf(sm[0], 1.0f);
}

// ---------------------------------------------------------------------------
// Kernel 9: x_att = attn @ v  (M=96, N=320, K=256), 4 accumulators/wave
// ---------------------------------------------------------------------------
__global__ __launch_bounds__(256) void k_avgemm(
    const bf16_t* attn_bf, const bf16_t* v_bf, float* x_att) {
  int wid = blockIdx.x * 8 + (threadIdx.x >> 5);
  if (wid >= BB * 6 * 5) return;
  int lane = threadIdx.x & 31, col = lane & 15, hi = lane >> 4;
  int b = wid / 30, r = wid % 30, mt = r / 5, ng = r % 5;
  const bf16_t* Arow = attn_bf + ((size_t)b * RP + mt * 16 + col) * PP;
  const bf16_t* Bcol = v_bf + ((size_t)b * CC + ng * 64 + col) * PP;
  v8f acc[4];
#pragma unroll
  for (int nt = 0; nt < 4; ++nt) acc[nt] = acc_zero();
#pragma unroll 2
  for (int k0 = 0; k0 < PP; k0 += 32) {
    v16bf a = loadA(Arow + k0, hi);
    v16bf bm[4];
#pragma unroll
    for (int nt = 0; nt < 4; ++nt)
      bm[nt] = loadB(Bcol + (size_t)nt * 16 * PP + k0, hi);
#pragma unroll
    for (int nt = 0; nt < 4; ++nt)
      acc[nt] = WMMA_BF16(a, bm[nt], acc[nt]);
  }
  int pbase = mt * 16 + hi * 8;
#pragma unroll
  for (int nt = 0; nt < 4; ++nt) {
    int o = ng * 64 + nt * 16 + col;
    float* dst = x_att + ((size_t)b * CC + o) * RP + pbase;
#pragma unroll
    for (int i = 0; i < 8; ++i) dst[i] = acc[nt][i];
  }
}

// ---------------------------------------------------------------------------
// Kernel 10: align_context = attn @ context (M=96, N=1024, K=256), 4 acc/wave
// ---------------------------------------------------------------------------
__global__ __launch_bounds__(256) void k_acgemm(
    const bf16_t* attn_bf, const bf16_t* ctxT_bf, bf16_t* ac_bf) {
  int wid = blockIdx.x * 8 + (threadIdx.x >> 5);
  if (wid >= BB * 6 * 16) return;
  int lane = threadIdx.x & 31, col = lane & 15, hi = lane >> 4;
  int b = wid / 96, r = wid % 96, mt = r / 16, ng = r % 16;
  const bf16_t* Arow = attn_bf + ((size_t)b * RP + mt * 16 + col) * PP;
  const bf16_t* Bcol = ctxT_bf + ((size_t)b * CTXC + ng * 64 + col) * PP;
  v8f acc[4];
#pragma unroll
  for (int nt = 0; nt < 4; ++nt) acc[nt] = acc_zero();
#pragma unroll 2
  for (int k0 = 0; k0 < PP; k0 += 32) {
    v16bf a = loadA(Arow + k0, hi);
    v16bf bm[4];
#pragma unroll
    for (int nt = 0; nt < 4; ++nt)
      bm[nt] = loadB(Bcol + (size_t)nt * 16 * PP + k0, hi);
#pragma unroll
    for (int nt = 0; nt < 4; ++nt)
      acc[nt] = WMMA_BF16(a, bm[nt], acc[nt]);
  }
#pragma unroll
  for (int nt = 0; nt < 4; ++nt)
#pragma unroll
    for (int i = 0; i < 8; ++i)
      ac_bf[((size_t)b * RP + mt * 16 + hi * 8 + i) * CTXC + ng * 64 + nt * 16 + col] =
          (bf16_t)acc[nt][i];
}

// ---------------------------------------------------------------------------
// Kernel 11: sg/sb = conv3x3(align_context, w_sg / w_sb), 9 shifted taps,
// wave = 1 m-tile x 2 n-tiles x {sg,sb} = 4 accumulators, A reused 4x.
// ---------------------------------------------------------------------------
__global__ __launch_bounds__(256) void k_spadegemm(
    const bf16_t* ac_bf, const bf16_t* wt_sg, const bf16_t* wt_sb,
    const float* b_sg, const float* b_sb, float* sg_f, float* sb_f) {
  int wid = blockIdx.x * 8 + (threadIdx.x >> 5);
  if (wid >= BB * 6 * 10) return;
  int lane = threadIdx.x & 31, col = lane & 15, hi = lane >> 4;
  int b = wid / 60, r = wid % 60, mt = r / 10, ng = r % 10;
  int p = mt * 16 + col;
  int y = p / 9, x = p % 9;
  v8f a1[2], a2[2];
#pragma unroll
  for (int nt = 0; nt < 2; ++nt) { a1[nt] = acc_zero(); a2[nt] = acc_zero(); }
  for (int t = 0; t < 9; ++t) {
    int dy = t / 3 - 1, dx = t % 3 - 1;
    int yy = y + dy, xx = x + dx;
    bool valid = (p < RR) && (yy >= 0) && (yy < 9) && (xx >= 0) && (xx < 9);
    const bf16_t* Arow = ac_bf + ((size_t)b * RP + yy * 9 + xx) * CTXC;
    const bf16_t* B1 = wt_sg + ((size_t)t * CC + ng * 32 + col) * CTXC;
    const bf16_t* B2 = wt_sb + ((size_t)t * CC + ng * 32 + col) * CTXC;
#pragma unroll 2
    for (int k0 = 0; k0 < CTXC; k0 += 32) {
      v16bf a = valid ? loadA(Arow + k0, hi) : frag_zero();
      v16bf bm1[2], bm2[2];
#pragma unroll
      for (int nt = 0; nt < 2; ++nt) {
        bm1[nt] = loadB(B1 + (size_t)nt * 16 * CTXC + k0, hi);
        bm2[nt] = loadB(B2 + (size_t)nt * 16 * CTXC + k0, hi);
      }
#pragma unroll
      for (int nt = 0; nt < 2; ++nt) {
        a1[nt] = WMMA_BF16(a, bm1[nt], a1[nt]);
        a2[nt] = WMMA_BF16(a, bm2[nt], a2[nt]);
      }
    }
  }
  int pbase = mt * 16 + hi * 8;
#pragma unroll
  for (int nt = 0; nt < 2; ++nt) {
    int o = ng * 32 + nt * 16 + col;
    float bg = b_sg[o], bb2 = b_sb[o];
    float* d1 = sg_f + ((size_t)b * CC + o) * RP + pbase;
    float* d2 = sb_f + ((size_t)b * CC + o) * RP + pbase;
#pragma unroll
    for (int i = 0; i < 8; ++i) { d1[i] = a1[nt][i] + bg; d2[i] = a2[nt][i] + bb2; }
  }
}

// ---------------------------------------------------------------------------
// Kernel 12: group_norm(x_att, eps 1e-5) * sg + sb, select by align_mask
// ---------------------------------------------------------------------------
__global__ __launch_bounds__(256) void k_gnspade(
    const float* x_att, const float* sg_f, const float* sb_f,
    const float* amask, float* x_final) {
  int b = blockIdx.x >> 5, g = blockIdx.x & 31, tid = threadIdx.x;
  __shared__ float s1[256], s2[256];
  float su = 0.f, sq = 0.f;
  for (int i = tid; i < 10 * RR; i += 256) {
    int c = g * 10 + i / RR, p = i % RR;
    float v = x_att[((size_t)b * CC + c) * RP + p];
    su += v; sq += v * v;
  }
  s1[tid] = su; s2[tid] = sq; __syncthreads();
  for (int off = 128; off > 0; off >>= 1) {
    if (tid < off) { s1[tid] += s1[tid + off]; s2[tid] += s2[tid + off]; }
    __syncthreads();
  }
  float mu = s1[0] / 810.f;
  float var = s2[0] / 810.f - mu * mu;
  float rs = rsqrtf(var + 1e-5f);
  for (int i = tid; i < 10 * RR; i += 256) {
    int c = g * 10 + i / RR, p = i % RR;
    size_t idx = ((size_t)b * CC + c) * RP + p;
    float xa = x_att[idx];
    float spade = (xa - mu) * rs * sg_f[idx] + sb_f[idx];
    float am = amask[b * RR + p];
    x_final[((size_t)b * CC + c) * RR + p] = (am > 0.5f) ? spade : xa;
  }
}

// ---------------------------------------------------------------------------
// Kernel 13: writeback (HBM-bound: stream global_x in/out once)
// ---------------------------------------------------------------------------
__global__ __launch_bounds__(256) void k_writeback(
    const float* gx, const float* bbox, const float* x_final, float* out) {
  int b = blockIdx.x >> 4;
  int pix = ((blockIdx.x & 15) << 8) + threadIdx.x;   // 0..4095
  int y = pix >> 6, x = pix & 63;
  int bx1 = (int)(bbox[b * 4 + 0] * 64.f);
  int by1 = (int)(bbox[b * 4 + 1] * 64.f);
  int bx2 = (int)(bbox[b * 4 + 2] * 64.f);
  int by2 = (int)(bbox[b * 4 + 3] * 64.f);
  bx2 = max(bx2, bx1 + 1); by2 = max(by2, by1 + 1);
  bool inside = (y >= by1) && (y < by2) && (x >= bx1) && (x < bx2);
  int pp = 0;
  if (inside) {
    int iy = min(max((y - by1) * 9 / (by2 - by1), 0), 8);
    int ix = min(max((x - bx1) * 9 / (bx2 - bx1), 0), 8);
    pp = iy * 9 + ix;
  }
  for (int c = 0; c < CC; ++c) {
    size_t gidx = ((size_t)b * CC + c) * HW + pix;
    float v = gx[gidx];
    if (inside) v += x_final[((size_t)b * CC + c) * RR + pp];
    out[gidx] = v;
  }
}

// ---------------------------------------------------------------------------
extern "C" void kernel_launch(void* const* d_in, const int* in_sizes, int n_in,
                              void* d_out, int out_size, void* d_ws, size_t ws_size,
                              hipStream_t stream) {
  (void)in_sizes; (void)n_in; (void)out_size; (void)ws_size;
  const float* global_x  = (const float*)d_in[0];
  const float* context   = (const float*)d_in[1];
  const float* indicator = (const float*)d_in[2];
  const float* bbox      = (const float*)d_in[3];
  const float* mask      = (const float*)d_in[4];
  const float* ln_gamma  = (const float*)d_in[5];
  const float* ln_beta   = (const float*)d_in[6];
  const float* cn_gamma  = (const float*)d_in[7];
  const float* cn_beta   = (const float*)d_in[8];
  const float* w_in      = (const float*)d_in[9];
  const float* b_in      = (const float*)d_in[10];
  const float* w_ctx     = (const float*)d_in[11];
  const float* b_ctx     = (const float*)d_in[12];
  const float* w_out     = (const float*)d_in[13];
  const float* b_out     = (const float*)d_in[14];
  const float* w_sg      = (const float*)d_in[15];
  const float* b_sg      = (const float*)d_in[16];
  const float* w_sb      = (const float*)d_in[17];
  const float* b_sb      = (const float*)d_in[18];

  float* out_gx   = (float*)d_out;
  float* out_mask = out_gx + (size_t)BB * CC * HW;

  char* ws = (char*)d_ws;
  size_t off = 0;
  auto alloc = [&](size_t bytes) -> void* {
    void* p = ws + off;
    off = (off + bytes + 255) & ~(size_t)255;
    return p;
  };
  bf16_t* w_ctx_bf = (bf16_t*)alloc((size_t)CC * CTXC * 2);
  bf16_t* wt_out   = (bf16_t*)alloc((size_t)9 * CC * CTXC * 2);
  bf16_t* wt_sg    = (bf16_t*)alloc((size_t)9 * CC * CTXC * 2);
  bf16_t* wt_sb    = (bf16_t*)alloc((size_t)9 * CC * CTXC * 2);
  bf16_t* ctx_bf   = (bf16_t*)alloc((size_t)BB * PP * CTXC * 2);
  bf16_t* ctxT_bf  = (bf16_t*)alloc((size_t)BB * CTXC * PP * 2);
  bf16_t* kn_bf    = (bf16_t*)alloc((size_t)BB * PP * CTXC * 2);
  float*  xroi     = (float*) alloc((size_t)BB * CC * RR * 4);
  float*  gnA      = (float*) alloc((size_t)BB * 32 * 2 * 4);
  bf16_t* q_bf     = (bf16_t*)alloc((size_t)BB * RP * CC * 2);
  bf16_t* k_bf     = (bf16_t*)alloc((size_t)BB * PP * CC * 2);
  bf16_t* v_bf     = (bf16_t*)alloc((size_t)BB * CC * PP * 2);
  float*  sim_f    = (float*) alloc((size_t)BB * RP * PP * 4);
  bf16_t* attn_bf  = (bf16_t*)alloc((size_t)BB * RP * PP * 2);
  bf16_t* ac_bf    = (bf16_t*)alloc((size_t)BB * RP * CTXC * 2);
  float*  x_att    = (float*) alloc((size_t)BB * CC * RP * 4);
  float*  sg_f     = (float*) alloc((size_t)BB * CC * RP * 4);
  float*  sb_f     = (float*) alloc((size_t)BB * CC * RP * 4);
  float*  x_final  = (float*) alloc((size_t)BB * CC * RR * 4);

  k_repack<<<1280, 256, 0, stream>>>(w_ctx, w_out, w_sg, w_sb,
                                     w_ctx_bf, wt_out, wt_sg, wt_sb);
  k_ctxnorm<<<BB * 32, 256, 0, stream>>>(context, cn_gamma, cn_beta,
                                         ctx_bf, ctxT_bf, kn_bf);
  k_roialign<<<BB * 102, 256, 0, stream>>>(global_x, bbox, xroi);
  k_lnstats<<<BB * 32, 256, 0, stream>>>(xroi, gnA);
  k_q<<<BB * 120, 256, 0, stream>>>(xroi, gnA, ln_gamma, ln_beta,
                                    w_in, b_in, indicator, q_bf);
  k_kgemm<<<(BB * 16 * 5) / 8, 256, 0, stream>>>(kn_bf, w_ctx_bf, b_ctx, k_bf);
  k_vgemm<<<BB * 5, 256, 0, stream>>>(ctx_bf, wt_out, b_out, v_bf);
  k_simgemm<<<(BB * 6 * 4) / 8, 256, 0, stream>>>(q_bf, k_bf, sim_f);
  k_softmax<<<BB * RP, 256, 0, stream>>>(sim_f, mask, attn_bf, out_mask);
  k_avgemm<<<(BB * 6 * 5) / 8, 256, 0, stream>>>(attn_bf, v_bf, x_att);
  k_acgemm<<<(BB * 6 * 16) / 8, 256, 0, stream>>>(attn_bf, ctxT_bf, ac_bf);
  k_spadegemm<<<(BB * 6 * 10) / 8, 256, 0, stream>>>(ac_bf, wt_sg, wt_sb,
                                                     b_sg, b_sb, sg_f, sb_f);
  k_gnspade<<<BB * 32, 256, 0, stream>>>(x_att, sg_f, sb_f, out_mask, x_final);
  k_writeback<<<BB * 16, 256, 0, stream>>>(global_x, bbox, x_final, out_gx);
}
